// MPNetSelfAttention_42906723287251
// MI455X (gfx1250) — matrix operand order repaired and checked by reading, not compiled, not run on hardware
//
#include <hip/hip_runtime.h>
#include <hip/hip_bf16.h>
#include <stdint.h>

typedef __attribute__((ext_vector_type(16))) _Float16 v16h;
typedef __attribute__((ext_vector_type(8)))  _Float16 v8h;
typedef __attribute__((ext_vector_type(8)))  float    v8f;

#define H_DIM 768
#define NHEAD 12
#define HDIM  64
#define BATCH 2
#define SEQ   2048

#define KTP 72   // LDS pitch (halves) for K tile  [32 k][64 d] (row-major, 144B rows)
#define VTP 40   // LDS pitch (halves) for V^T tile [64 d][32 k]
#define PP  40   // LDS pitch (halves) for per-wave P tile [16 m][32 k]

// ---------------- WMMA operand loaders (16-bit, 16x16x32 shape) ----------------
// A-matrix 16x32 (MxK), per ISA 7.12.2: lane L -> M = L%16, hs = L/16,
// halves[0..7] = A[M][hs*8 + 0..7], halves[8..15] = A[M][16 + hs*8 + 0..7]
__device__ __forceinline__ v16h load_A16(const _Float16* __restrict__ p, int ld, int m0, int k0) {
  const int lane = threadIdx.x & 31;
  const int m = m0 + (lane & 15);
  const int hs = lane >> 4;
  const _Float16* r = p + (size_t)m * ld + k0 + hs * 8;
  v8h lo = *(const v8h*)(r);
  v8h hi = *(const v8h*)(r + 16);
  v16h a;
#pragma unroll
  for (int i = 0; i < 8; ++i) { a[i] = lo[i]; a[8 + i] = hi[i]; }
  return a;
}

// B-matrix 32x16 (KxN): lane L -> N = L%16, halves[0..7] = B[(L/16)*8 + 0..7][N], halves[8..15] = B[16 + (L/16)*8 + 0..7][N].
// Column N's K-run must be contiguous at p + N*ld + k.
__device__ __forceinline__ v16h load_B16(const _Float16* __restrict__ p, int ld, int k0, int n0) {
  const int lane = threadIdx.x & 31;
  const int n = n0 + (lane & 15);
  const int hs = lane >> 4;
  const _Float16* c = p + (size_t)n * ld + k0 + hs * 8;
  v8h lo = *(const v8h*)(c);
  v8h hi = *(const v8h*)(c + 16);
  v16h b;
#pragma unroll
  for (int i = 0; i < 8; ++i) { b[i] = lo[i]; b[8 + i] = hi[i]; }
  return b;
}

// CDNA5 async copy global->LDS (16B), tracked by ASYNCcnt
__device__ __forceinline__ void async_ld_b128(const _Float16* lds_dst, const _Float16* gsrc) {
  unsigned ldsoff = (unsigned)(uintptr_t)lds_dst;   // low 32 bits = LDS offset
  asm volatile("global_load_async_to_lds_b128 %0, %1, off"
               :: "v"(ldsoff), "v"(gsrc) : "memory");
}

// ---------------- fp32 -> f16 conversion ----------------
__global__ void cvt_f32_to_f16(const float* __restrict__ in, _Float16* __restrict__ out, size_t n) {
  size_t i = (size_t)blockIdx.x * blockDim.x + threadIdx.x;
  size_t stride = (size_t)gridDim.x * blockDim.x;
  for (; i < n; i += stride) out[i] = (_Float16)in[i];
}

// ---------------- GEMM: out = A(f16)[M,K] @ W(f16)[N,K]^T + bias ----------------
// 128 threads = 4 waves, tile 64(M) x 64(N), wave owns 16(M) x 64(N).
// Software-pipelined: operands for k+32 are loaded before the WMMAs for k.
__global__ __launch_bounds__(128) void wmma_gemm_xwT(
    const _Float16* __restrict__ A, const _Float16* __restrict__ W,
    const float* __restrict__ bias, _Float16* __restrict__ outh,
    float* __restrict__ outf, int M, int N, int K)
{
  const int wave = threadIdx.x >> 5;
  const int lane = threadIdx.x & 31;
  const int ln = lane & 15;
  const int hs = lane >> 4;
  const int m0 = blockIdx.x * 64 + wave * 16;
  const int n0 = blockIdx.y * 64;

  v8f acc[4] = {};
  v16h a  = load_A16(A, K, m0, 0);
  v16h b0 = load_B16(W, K, 0, n0);
  v16h b1 = load_B16(W, K, 0, n0 + 16);
  v16h b2 = load_B16(W, K, 0, n0 + 32);
  v16h b3 = load_B16(W, K, 0, n0 + 48);

  for (int kk = 0; kk < K; kk += 32) {
    const int kn = (kk + 32 < K) ? kk + 32 : 0;   // last-iter reload is harmless
    v16h an  = load_A16(A, K, m0, kn);
    v16h bn0 = load_B16(W, K, kn, n0);
    v16h bn1 = load_B16(W, K, kn, n0 + 16);
    v16h bn2 = load_B16(W, K, kn, n0 + 32);
    v16h bn3 = load_B16(W, K, kn, n0 + 48);
    acc[0] = __builtin_amdgcn_wmma_f32_16x16x32_f16(false, a, false, b0, (short)0, acc[0], false, false);
    acc[1] = __builtin_amdgcn_wmma_f32_16x16x32_f16(false, a, false, b1, (short)0, acc[1], false, false);
    acc[2] = __builtin_amdgcn_wmma_f32_16x16x32_f16(false, a, false, b2, (short)0, acc[2], false, false);
    acc[3] = __builtin_amdgcn_wmma_f32_16x16x32_f16(false, a, false, b3, (short)0, acc[3], false, false);
    a = an; b0 = bn0; b1 = bn1; b2 = bn2; b3 = bn3;
  }
#pragma unroll
  for (int t = 0; t < 4; ++t) {
    const int n = n0 + t * 16 + ln;
    const float bv = bias ? bias[n] : 0.f;
#pragma unroll
    for (int r = 0; r < 8; ++r) {
      const int m = m0 + r + 8 * hs;             // C/D layout: VGPR r, half-wave -> M
      const float v = acc[t][r] + bv;
      if (outh) outh[(size_t)m * N + n] = (_Float16)v;
      else      outf[(size_t)m * N + n] = v;
    }
  }
}

// ---------------- fused flash attention ----------------
// grid = (B*NH, SEQ/64), block = 128 (4 waves); wave handles 16 query rows.
// K/V tiles double-buffered in LDS; K staged with async-to-LDS copies.
__global__ __launch_bounds__(128) void wmma_flash_attn(
    const _Float16* __restrict__ Qh, const _Float16* __restrict__ Kh,
    const _Float16* __restrict__ Vh, const float* __restrict__ pbias,
    const float* __restrict__ amask, _Float16* __restrict__ Ch)
{
  __shared__ _Float16 KT[2][32 * KTP];   // K tile row-major [key][d] (B-layout ready)
  __shared__ _Float16 VT[2][64 * VTP];   // V^T tile [d][key]
  __shared__ _Float16 PT[4][16 * PP];    // per-wave probabilities [m][key]

  const int tid = threadIdx.x;
  const int wave = tid >> 5;
  const int lane = tid & 31;
  const int ln = lane & 15;
  const int hs = lane >> 4;
  const int bh = blockIdx.x;
  const int b = bh / NHEAD;
  const int h = bh % NHEAD;
  const int q0 = blockIdx.y * 64 + wave * 16;
  const size_t seqbase = (size_t)b * SEQ * H_DIM;
  const int hoff = h * HDIM;

  const _Float16* qbase = Qh + seqbase + hoff;
  const _Float16* kbase = Kh + seqbase + hoff;
  const float* pb = pbias + (size_t)h * SEQ * SEQ;
  const float* mk = amask + (size_t)b * SEQ;

  // Q tile (16x64) in A-layout, kept in registers for the whole pass
  v16h qa0 = load_A16(qbase, H_DIM, q0, 0);
  v16h qa1 = load_A16(qbase, H_DIM, q0, 32);

  v8f o[4] = {};
  float mrow[8], lrow[8];
#pragma unroll
  for (int r = 0; r < 8; ++r) { mrow[r] = -1e30f; lrow[r] = 0.f; }

  // stage K (async copies) + V (register transpose) tile for keys [kt, kt+32)
  auto stage = [&](int kt, int buf) {
    // K: 32 rows x 128B = 256 x 16B chunks, 2 per thread; LDS rows padded to 144B
#pragma unroll
    for (int i = 0; i < 2; ++i) {
      const int c = tid * 2 + i;
      const int row = c >> 3;          // 0..31 (key within tile)
      const int seg = c & 7;           // 0..7  (8-half segment of d)
      async_ld_b128(&KT[buf][row * KTP + seg * 8],
                    kbase + (size_t)(kt + row) * H_DIM + seg * 8);
    }
    // V: thread moves 16 d-values of one key row into transposed LDS tile
    const int k = kt + (tid >> 2);
    const int d0 = (tid & 3) * 16;
    const _Float16* vrow = Vh + seqbase + (size_t)k * H_DIM + hoff + d0;
    v8h va = *(const v8h*)(vrow);
    v8h vc = *(const v8h*)(vrow + 8);
#pragma unroll
    for (int j = 0; j < 8; ++j) {
      VT[buf][(d0 + j) * VTP + (k - kt)]     = va[j];
      VT[buf][(d0 + 8 + j) * VTP + (k - kt)] = vc[j];
    }
  };

  stage(0, 0);
  asm volatile("s_wait_asynccnt 0" ::: "memory");
  __syncthreads();
  int cur = 0;

  for (int kt = 0; kt < SEQ; kt += 32) {
    if (kt + 32 < SEQ) stage(kt + 32, cur ^ 1);   // overlap staging with compute

    // position-bias / mask streaming (roofline-dominant): issue before WMMAs
    const float mk0 = mk[kt + ln];
    const float mk1 = mk[kt + 16 + ln];
    float pb0[8], pb1[8];
#pragma unroll
    for (int r = 0; r < 8; ++r) {
      const int q = q0 + r + 8 * hs;
      pb0[r] = pb[(size_t)q * SEQ + kt + ln];
      pb1[r] = pb[(size_t)q * SEQ + kt + 16 + ln];
      if (kt + 32 < SEQ) __builtin_prefetch(pb + (size_t)q * SEQ + kt + 32 + ln, 0, 1);
    }

    // scores S[16 x 32] = Q . K^T  (K tile rows in LDS are B columns)
    const _Float16* KTc = &KT[cur][0];
    v16h kb00 = load_B16(KTc, KTP, 0,  0);
    v16h kb01 = load_B16(KTc, KTP, 32, 0);
    v16h kb10 = load_B16(KTc, KTP, 0,  16);
    v16h kb11 = load_B16(KTc, KTP, 32, 16);
    v8f s0 = {}, s1 = {};
    s0 = __builtin_amdgcn_wmma_f32_16x16x32_f16(false, qa0, false, kb00, (short)0, s0, false, false);
    s0 = __builtin_amdgcn_wmma_f32_16x16x32_f16(false, qa1, false, kb01, (short)0, s0, false, false);
    s1 = __builtin_amdgcn_wmma_f32_16x16x32_f16(false, qa0, false, kb10, (short)0, s1, false, false);
    s1 = __builtin_amdgcn_wmma_f32_16x16x32_f16(false, qa1, false, kb11, (short)0, s1, false, false);

    // online softmax (C-layout: row M = r + 8*hs, col = lane%16)
    float p0[8], p1[8];
#pragma unroll
    for (int r = 0; r < 8; ++r) {
      const float x0 = s0[r] * 0.125f + pb0[r] + mk0;
      const float x1 = s1[r] * 0.125f + pb1[r] + mk1;
      float mx = fmaxf(x0, x1);
      mx = fmaxf(mx, __shfl_xor(mx, 1, 32));
      mx = fmaxf(mx, __shfl_xor(mx, 2, 32));
      mx = fmaxf(mx, __shfl_xor(mx, 4, 32));
      mx = fmaxf(mx, __shfl_xor(mx, 8, 32));
      const float mnew  = fmaxf(mrow[r], mx);
      const float alpha = __expf(mrow[r] - mnew);
      const float e0 = __expf(x0 - mnew);
      const float e1 = __expf(x1 - mnew);
      float rs = e0 + e1;
      rs += __shfl_xor(rs, 1, 32);
      rs += __shfl_xor(rs, 2, 32);
      rs += __shfl_xor(rs, 4, 32);
      rs += __shfl_xor(rs, 8, 32);
      lrow[r] = lrow[r] * alpha + rs;
      mrow[r] = mnew;
#pragma unroll
      for (int t = 0; t < 4; ++t) o[t][r] *= alpha;
      p0[r] = e0; p1[r] = e1;
    }

    // C-layout -> A-layout for P via per-wave LDS tile (DS in-order per wave)
    _Float16* Pw = &PT[wave][0];
#pragma unroll
    for (int r = 0; r < 8; ++r) {
      Pw[(r + 8 * hs) * PP + ln]      = (_Float16)p0[r];
      Pw[(r + 8 * hs) * PP + 16 + ln] = (_Float16)p1[r];
    }
    asm volatile("s_wait_dscnt 0" ::: "memory");
    v16h pa = load_A16(Pw, PP, 0, 0);

    // O[16 x 64] += P(16x32) . V(32x64); V^T rows are B columns
    const _Float16* VTc = &VT[cur][0];
#pragma unroll
    for (int t = 0; t < 4; ++t) {
      v16h vb = load_B16(VTc, VTP, 0, t * 16);
      o[t] = __builtin_amdgcn_wmma_f32_16x16x32_f16(false, pa, false, vb,
                                                    (short)0, o[t], false, false);
    }

    // publish next buffers: own async K copies must land, then barrier
    asm volatile("s_wait_asynccnt 0" ::: "memory");
    __syncthreads();
    cur ^= 1;
  }

  // normalize and store context (f16) in [B*S, H] layout
#pragma unroll
  for (int r = 0; r < 8; ++r) {
    const float inv = 1.f / lrow[r];
    const int q = q0 + r + 8 * hs;
#pragma unroll
    for (int t = 0; t < 4; ++t) {
      Ch[seqbase + (size_t)q * H_DIM + hoff + t * 16 + ln] = (_Float16)(o[t][r] * inv);
    }
  }
}

// ---------------- host launcher ----------------
extern "C" void kernel_launch(void* const* d_in, const int* in_sizes, int n_in,
                              void* d_out, int out_size, void* d_ws, size_t ws_size,
                              hipStream_t stream) {
  (void)in_sizes; (void)n_in; (void)out_size; (void)ws_size;
  const float* hidden = (const float*)d_in[0];
  const float* amask  = (const float*)d_in[1];
  const float* pbias  = (const float*)d_in[2];
  const float* Wq = (const float*)d_in[3];
  const float* bq = (const float*)d_in[4];
  const float* Wk = (const float*)d_in[5];
  const float* bk = (const float*)d_in[6];
  const float* Wv = (const float*)d_in[7];
  const float* bv = (const float*)d_in[8];
  const float* Wo = (const float*)d_in[9];
  const float* bo = (const float*)d_in[10];
  float* out = (float*)d_out;

  const size_t MS = (size_t)BATCH * SEQ;     // 4096 rows
  const size_t XN = MS * H_DIM;              // activation elems
  const size_t WN = (size_t)H_DIM * H_DIM;   // weight elems

  _Float16* base = (_Float16*)d_ws;
  _Float16* Xh  = base; base += XN;
  _Float16* Wqh = base; base += WN;
  _Float16* Wkh = base; base += WN;
  _Float16* Wvh = base; base += WN;
  _Float16* Woh = base; base += WN;
  _Float16* Qh  = base; base += XN;
  _Float16* Kh  = base; base += XN;
  _Float16* Vh  = base; base += XN;
  _Float16* Ch  = base; base += XN;

  const int thr = 256;
  cvt_f32_to_f16<<<(int)((XN + thr * 4 - 1) / (thr * 4)), thr, 0, stream>>>(hidden, Xh, XN);
  cvt_f32_to_f16<<<(int)((WN + thr * 4 - 1) / (thr * 4)), thr, 0, stream>>>(Wq, Wqh, WN);
  cvt_f32_to_f16<<<(int)((WN + thr * 4 - 1) / (thr * 4)), thr, 0, stream>>>(Wk, Wkh, WN);
  cvt_f32_to_f16<<<(int)((WN + thr * 4 - 1) / (thr * 4)), thr, 0, stream>>>(Wv, Wvh, WN);
  cvt_f32_to_f16<<<(int)((WN + thr * 4 - 1) / (thr * 4)), thr, 0, stream>>>(Wo, Woh, WN);

  dim3 gg((unsigned)(MS / 64), H_DIM / 64);  // (64, 12)
  wmma_gemm_xwT<<<gg, 128, 0, stream>>>(Xh, Wqh, bq, Qh, nullptr, (int)MS, H_DIM, H_DIM);
  wmma_gemm_xwT<<<gg, 128, 0, stream>>>(Xh, Wkh, bk, Kh, nullptr, (int)MS, H_DIM, H_DIM);
  wmma_gemm_xwT<<<gg, 128, 0, stream>>>(Xh, Wvh, bv, Vh, nullptr, (int)MS, H_DIM, H_DIM);

  wmma_flash_attn<<<dim3(BATCH * NHEAD, SEQ / 64), 128, 0, stream>>>(Qh, Kh, Vh, pbias, amask, Ch);

  wmma_gemm_xwT<<<gg, 128, 0, stream>>>(Ch, Woh, bo, nullptr, out, (int)MS, H_DIM, H_DIM);
}
